// MultiHeadSelfAttention_39187281609243
// MI455X (gfx1250) — matrix operand order repaired and checked by reading, not compiled, not run on hardware
//
#include <hip/hip_runtime.h>

// MHA: B=2, T=4096, D_MODEL=768, H=12, d_head=64. fp32 in/out, bf16 WMMA internally.
#define NHEADS 12
#define TSEQ   4096
#define DMODEL 768
#define DHEAD  64
#define NROWS  8192   // B*T

typedef __attribute__((ext_vector_type(16))) __bf16          v16bf;
typedef __attribute__((ext_vector_type(8)))  float           v8f;
typedef __attribute__((ext_vector_type(4)))  float           vf4;
typedef __attribute__((ext_vector_type(8)))  unsigned short  v8u;
typedef __attribute__((ext_vector_type(16))) unsigned short  v16u;
typedef __attribute__((ext_vector_type(4)))  unsigned short  v4u;

// GCC-style vector to match the async-to-LDS builtin's parameter type
typedef int v4i __attribute__((vector_size(16)));
typedef v4i __attribute__((address_space(1)))* g_v4i_ptr;   // global
typedef v4i __attribute__((address_space(3)))* l_v4i_ptr;   // LDS

#if defined(__has_builtin)
#if __has_builtin(__builtin_amdgcn_global_load_async_to_lds_b128)
#define HAVE_ASYNC_LDS 1
#endif
#endif
#ifndef HAVE_ASYNC_LDS
#define HAVE_ASYNC_LDS 0
#endif

static __device__ __forceinline__ unsigned short f2bf(float f) {
  unsigned int u = __float_as_uint(f);
  u += 0x7FFFu + ((u >> 16) & 1u);          // round-to-nearest-even
  return (unsigned short)(u >> 16);
}

static __device__ __forceinline__ v16bf frag_cat(v8u lo, v8u hi) {
  v16u c = __builtin_shufflevector(lo, hi, 0,1,2,3,4,5,6,7,8,9,10,11,12,13,14,15);
  union { v16u u; v16bf b; } x;
  x.u = c;
  return x.b;
}

static __device__ __forceinline__ v8f wmma_bf16(v16bf a, v16bf b, v8f c) {
  return __builtin_amdgcn_wmma_f32_16x16x32_bf16(false, a, false, b, (short)0, c,
                                                 false, false);
}

// 16-byte global -> LDS copy; async (ASYNCcnt) path on CDNA5 when available.
static __device__ __forceinline__ void copy16_g2lds(const void* gsrc, void* ldst) {
#if HAVE_ASYNC_LDS
  __builtin_amdgcn_global_load_async_to_lds_b128((g_v4i_ptr)gsrc, (l_v4i_ptr)ldst,
                                                 0, 0);
#else
  *(v8u*)ldst = *(const v8u*)gsrc;
#endif
}

static __device__ __forceinline__ void async_fence() {
#if HAVE_ASYNC_LDS
#if __has_builtin(__builtin_amdgcn_s_wait_asynccnt)
  __builtin_amdgcn_s_wait_asynccnt(0);
#else
  asm volatile("s_wait_asynccnt 0" ::: "memory");
#endif
#endif
}

// ---------------------------------------------------------------------------
// GEMM: Y[row, col] = sum_k A[row,k] * W[col,k] + bias[col]
//   A: [NROWS, 768] (f32 or bf16-as-u16), W: [768,768] f32.
//   OUT_HEADMAJOR=1: bf16 out at [B,H,T,64];  =0: f32 out at [NROWS,768].
// Block 256 threads = 8 waves (4 over M x 2 over N), tile 128x128, K-step 32.
// Per wave: 32x64 = 2x4 WMMA tiles -> 8 WMMAs per 6 fragment loads.
// ---------------------------------------------------------------------------
template <int IN_BF16, int OUT_HEADMAJOR>
__global__ __launch_bounds__(256) void gemm_wmma(const void* __restrict__ Ap,
                                                 const float* __restrict__ W,
                                                 const float* __restrict__ bias,
                                                 void* __restrict__ Yp) {
  constexpr int BM = 128, BN = 128, LDT = 40;  // 40 u16 = 80B row stride (16B mult)
  __shared__ unsigned short Xs[BM * LDT];
  __shared__ unsigned short Ws[BN * LDT];
  const int tid = threadIdx.x;
  const int lane = tid & 31, wave = tid >> 5;
  const int wm = wave & 3, wn = wave >> 2;     // 4 x 2 wave grid
  const int ln = lane & 15, hi = lane >> 4;
  const int M0 = blockIdx.x * BM, N0 = blockIdx.y * BN;

  const v8f vzero = {0.f, 0.f, 0.f, 0.f, 0.f, 0.f, 0.f, 0.f};
  v8f acc[2][4];
#pragma unroll
  for (int tm = 0; tm < 2; ++tm)
#pragma unroll
    for (int tn = 0; tn < 4; ++tn) acc[tm][tn] = vzero;

  for (int k0 = 0; k0 < DMODEL; k0 += 32) {
    __syncthreads();
    if constexpr (IN_BF16) {
      const unsigned short* A = (const unsigned short*)Ap;
#pragma unroll
      for (int p = 0; p < 2; ++p) {                 // 512 x 16B = 128x32 u16
        int idx = p * 256 + tid;
        int row = idx >> 2, c8 = (idx & 3) * 8;
        copy16_g2lds(&A[(size_t)(M0 + row) * DMODEL + k0 + c8],
                     &Xs[row * LDT + c8]);
      }
    } else {
      const float* A = (const float*)Ap;
#pragma unroll
      for (int p = 0; p < 4; ++p) {                 // 1024 x float4 = 128x32 f32
        int idx = p * 256 + tid;
        int row = idx >> 3, c4 = (idx & 7) * 4;
        vf4 v = *(const vf4*)&A[(size_t)(M0 + row) * DMODEL + k0 + c4];
        v4u t = {f2bf(v[0]), f2bf(v[1]), f2bf(v[2]), f2bf(v[3])};
        *(v4u*)&Xs[row * LDT + c4] = t;
      }
    }
#pragma unroll
    for (int p = 0; p < 4; ++p) {                   // 1024 x float4 = 128x32 f32
      int idx = p * 256 + tid;
      int row = idx >> 3, c4 = (idx & 7) * 4;
      vf4 v = *(const vf4*)&W[(size_t)(N0 + row) * DMODEL + k0 + c4];
      v4u t = {f2bf(v[0]), f2bf(v[1]), f2bf(v[2]), f2bf(v[3])};
      *(v4u*)&Ws[row * LDT + c4] = t;
    }
    if constexpr (IN_BF16) async_fence();
    __syncthreads();

    v16bf Af[2], Bf[4];
    const int ab = hi * 8, kb = hi * 16;
#pragma unroll
    for (int tm = 0; tm < 2; ++tm) {
      const unsigned short* r = &Xs[(wm * 32 + tm * 16 + ln) * LDT];
      Af[tm] = frag_cat(*(const v8u*)(r + ab), *(const v8u*)(r + 16 + ab));
    }
#pragma unroll
    for (int tn = 0; tn < 4; ++tn) {
      const unsigned short* r = &Ws[(wn * 64 + tn * 16 + ln) * LDT];
      Bf[tn] = frag_cat(*(const v8u*)(r + ab), *(const v8u*)(r + 16 + ab));
    }
#pragma unroll
    for (int tm = 0; tm < 2; ++tm)
#pragma unroll
      for (int tn = 0; tn < 4; ++tn)
        acc[tm][tn] = wmma_bf16(Af[tm], Bf[tn], acc[tm][tn]);
  }

#pragma unroll
  for (int tm = 0; tm < 2; ++tm)
#pragma unroll
    for (int tn = 0; tn < 4; ++tn)
#pragma unroll
      for (int r = 0; r < 8; ++r) {
        int row = M0 + wm * 32 + tm * 16 + hi * 8 + r;
        int col = N0 + wn * 64 + tn * 16 + ln;
        float val = acc[tm][tn][r] + bias[col];
        if constexpr (OUT_HEADMAJOR) {
          int b = row >> 12, t = row & (TSEQ - 1);
          int h = col >> 6, d = col & 63;
          ((unsigned short*)Yp)[(((size_t)(b * NHEADS + h)) * TSEQ + t) * DHEAD + d] =
              f2bf(val);
        } else {
          ((float*)Yp)[(size_t)row * DMODEL + col] = val;
        }
      }
}

// ---------------------------------------------------------------------------
// Flash attention, per (b,h). 128 threads = 4 waves; each wave owns 16 query
// rows; block handles 64 queries and streams 64-key K/V tiles through LDS.
// Q/K/V bf16 at [B,H,T,64]; output bf16 merged [B,T,768].
// ---------------------------------------------------------------------------
__global__ __launch_bounds__(128) void attn_kernel(const unsigned short* __restrict__ Qb,
                                                   const unsigned short* __restrict__ Kb,
                                                   const unsigned short* __restrict__ Vb,
                                                   unsigned short* __restrict__ Ob) {
  constexpr int LD = 72;  // 64+8 u16 -> 144B row stride (16B multiple)
  __shared__ unsigned short Qs[64 * LD];
  __shared__ unsigned short Ks[64 * LD];
  __shared__ unsigned short Vt[64 * LD];  // transposed: [d][key]
  __shared__ unsigned short Ps[64 * LD];

  const int qb = blockIdx.x;   // query block of 64 rows
  const int bh = blockIdx.y;   // b*NHEADS + h
  const int tid = threadIdx.x;
  const int lane = tid & 31, wave = tid >> 5;
  const int ln = lane & 15, hi = lane >> 4;
  const size_t base = (size_t)bh * TSEQ * DHEAD;

  // stage Q block (64x64 bf16) -- straight copy: use CDNA5 async-to-LDS
#pragma unroll
  for (int p = 0; p < 4; ++p) {
    int idx = p * 128 + tid;               // 512 x 16B
    int row = idx >> 3, c8 = (idx & 7) * 8;
    copy16_g2lds(&Qb[base + (size_t)(qb * 64 + row) * DHEAD + c8],
                 &Qs[row * LD + c8]);
  }
  async_fence();
  __syncthreads();

  v16bf Qf[2];
  {
    const unsigned short* r = &Qs[(wave * 16 + ln) * LD];
    const int ab = hi * 8;
    Qf[0] = frag_cat(*(const v8u*)(r + ab),      *(const v8u*)(r + 16 + ab));
    Qf[1] = frag_cat(*(const v8u*)(r + 32 + ab), *(const v8u*)(r + 48 + ab));
  }

  const v8f vzero = {0.f, 0.f, 0.f, 0.f, 0.f, 0.f, 0.f, 0.f};
  v8f O[4];
  O[0] = vzero; O[1] = vzero; O[2] = vzero; O[3] = vzero;
  float mrow[8], lrow[8];
#pragma unroll
  for (int r = 0; r < 8; ++r) { mrow[r] = -1e30f; lrow[r] = 0.f; }
  const float scale = 0.125f;  // 1/sqrt(64)

  for (int j = 0; j < TSEQ / 64; ++j) {
    __syncthreads();  // protect Ks/Vt from previous iteration's readers
    // stage K (async straight copy) and V (transposed, via VGPRs)
#pragma unroll
    for (int p = 0; p < 4; ++p) {
      int idx = p * 128 + tid;
      int row = idx >> 3, c8 = (idx & 7) * 8;
      copy16_g2lds(&Kb[base + (size_t)(j * 64 + row) * DHEAD + c8],
                   &Ks[row * LD + c8]);
      v8u vv = *(const v8u*)&Vb[base + (size_t)(j * 64 + row) * DHEAD + c8];
#pragma unroll
      for (int e = 0; e < 8; ++e) Vt[(c8 + e) * LD + row] = vv[e];
    }
    async_fence();
    __syncthreads();

    // S = Q * K^T  (4 tiles of 16x16, contraction over d=64)
    v8f S[4];
#pragma unroll
    for (int nt = 0; nt < 4; ++nt) {
      const unsigned short* r = &Ks[(nt * 16 + ln) * LD];
      const int kb = hi * 8;
      v16bf B0 = frag_cat(*(const v8u*)(r + kb),      *(const v8u*)(r + 16 + kb));
      v16bf B1 = frag_cat(*(const v8u*)(r + 32 + kb), *(const v8u*)(r + 48 + kb));
      v8f a = vzero;
      a = wmma_bf16(Qf[0], B0, a);
      a = wmma_bf16(Qf[1], B1, a);
      S[nt] = a;
    }

    // online softmax per row (row group = 16 lanes of this half-wave)
#pragma unroll
    for (int r = 0; r < 8; ++r) {
      float s0 = S[0][r] * scale, s1 = S[1][r] * scale;
      float s2 = S[2][r] * scale, s3 = S[3][r] * scale;
      float mx = fmaxf(fmaxf(s0, s1), fmaxf(s2, s3));
#pragma unroll
      for (int msk = 8; msk >= 1; msk >>= 1)
        mx = fmaxf(mx, __shfl_xor(mx, msk, 32));
      float mnew = fmaxf(mrow[r], mx);
      float corr = __expf(mrow[r] - mnew);
      mrow[r] = mnew;
      lrow[r] *= corr;
      O[0][r] *= corr; O[1][r] *= corr; O[2][r] *= corr; O[3][r] *= corr;
      float p0 = __expf(s0 - mnew), p1 = __expf(s1 - mnew);
      float p2 = __expf(s2 - mnew), p3 = __expf(s3 - mnew);
      int prow = (wave * 16 + hi * 8 + r) * LD;
      Ps[prow +  0 + ln] = f2bf(p0);
      Ps[prow + 16 + ln] = f2bf(p1);
      Ps[prow + 32 + ln] = f2bf(p2);
      Ps[prow + 48 + ln] = f2bf(p3);
      float ps = (p0 + p1) + (p2 + p3);
#pragma unroll
      for (int msk = 8; msk >= 1; msk >>= 1)
        ps += __shfl_xor(ps, msk, 32);
      lrow[r] += ps;
    }
    __syncthreads();  // C-layout -> A-layout redistribution via LDS

    // O += P * V   (V^T staged so B-fragment reads are contiguous b128)
    v16bf Pf[2];
    {
      const unsigned short* r = &Ps[(wave * 16 + ln) * LD];
      const int ab = hi * 8;
      Pf[0] = frag_cat(*(const v8u*)(r + ab),      *(const v8u*)(r + 16 + ab));
      Pf[1] = frag_cat(*(const v8u*)(r + 32 + ab), *(const v8u*)(r + 48 + ab));
    }
#pragma unroll
    for (int dt = 0; dt < 4; ++dt) {
      const unsigned short* r = &Vt[(dt * 16 + ln) * LD];
      const int kb = hi * 8;
      v16bf B0 = frag_cat(*(const v8u*)(r + kb),      *(const v8u*)(r + 16 + kb));
      v16bf B1 = frag_cat(*(const v8u*)(r + 32 + kb), *(const v8u*)(r + 48 + kb));
      O[dt] = wmma_bf16(Pf[0], B0, O[dt]);
      O[dt] = wmma_bf16(Pf[1], B1, O[dt]);
    }
  }

  // epilogue: normalize and write merged [B,T,768] bf16
  const int b = bh / NHEADS, h = bh % NHEADS;
#pragma unroll
  for (int dt = 0; dt < 4; ++dt)
#pragma unroll
    for (int r = 0; r < 8; ++r) {
      int t = qb * 64 + wave * 16 + hi * 8 + r;
      int d = dt * 16 + ln;
      float val = O[dt][r] / lrow[r];
      Ob[((size_t)(b * TSEQ + t)) * DMODEL + h * DHEAD + d] = f2bf(val);
    }
}

// ---------------------------------------------------------------------------
extern "C" void kernel_launch(void* const* d_in, const int* in_sizes, int n_in,
                              void* d_out, int out_size, void* d_ws, size_t ws_size,
                              hipStream_t stream) {
  (void)in_sizes; (void)n_in; (void)out_size; (void)ws_size;
  const float* x  = (const float*)d_in[0];
  const float* Wq = (const float*)d_in[1];
  const float* bq = (const float*)d_in[2];
  const float* Wk = (const float*)d_in[3];
  const float* bk = (const float*)d_in[4];
  const float* Wv = (const float*)d_in[5];
  const float* bv = (const float*)d_in[6];
  const float* Wo = (const float*)d_in[7];
  const float* bo = (const float*)d_in[8];

  const size_t SZ = (size_t)NROWS * DMODEL;  // elements per bf16 buffer
  unsigned short* Qb = (unsigned short*)d_ws;
  unsigned short* Kb = Qb + SZ;
  unsigned short* Vb = Kb + SZ;
  unsigned short* Ob = Vb + SZ;

  dim3 gg(NROWS / 128, DMODEL / 128);  // (64, 6)
  gemm_wmma<0, 1><<<gg, 256, 0, stream>>>(x, Wq, bq, Qb);
  gemm_wmma<0, 1><<<gg, 256, 0, stream>>>(x, Wk, bk, Kb);
  gemm_wmma<0, 1><<<gg, 256, 0, stream>>>(x, Wv, bv, Vb);

  attn_kernel<<<dim3(TSEQ / 64, 2 * NHEADS), 128, 0, stream>>>(Qb, Kb, Vb, Ob);

  gemm_wmma<1, 0><<<gg, 256, 0, stream>>>(Ob, Wo, bo, (float*)d_out);
}
